// ResSAGE_17970143167249
// MI455X (gfx1250) — matrix operand order repaired and checked by reading, not compiled, not run on hardware
//
#include <hip/hip_runtime.h>

#define H 128
#define LDSW 136            // 128 + 8 bf16 pad -> 272B row stride, conflict-free LDS banks
#define LRELU_SLOPE 0.01f
#define BN_EPS 1e-5f

typedef __attribute__((ext_vector_type(16))) __bf16 v16bf;
typedef __attribute__((ext_vector_type(8)))  __bf16 v8bf;
typedef __attribute__((ext_vector_type(8)))  float  v8f;
typedef __attribute__((ext_vector_type(4)))  unsigned int tdm_v4u;
typedef __attribute__((ext_vector_type(8)))  int          tdm_v8i;
typedef __attribute__((ext_vector_type(4)))  int          tdm_v4i;

static __device__ __forceinline__ unsigned short f2bf(float f) {
  // round-to-nearest-even fp32 -> bf16
  unsigned u = __float_as_uint(f);
  unsigned r = (u + 0x7FFFu + ((u >> 16) & 1u)) >> 16;
  return (unsigned short)r;
}

#if __has_builtin(__builtin_amdgcn_tensor_load_to_lds)
// TDM D#: 2D tensor 128x128 x 2B, tile = whole tensor, LDS padding:
// pad_interval=5 (64 DWORDs = one 256B row), pad_amount=3 (4 DWORDs = 16B)
static __device__ __forceinline__ void tdm_load_weight(const unsigned short* Wt,
                                                       unsigned ldsOff) {
  unsigned long long ga = (unsigned long long)Wt;
  tdm_v4u g0;
  g0[0] = 1u;                                        // count=1 (valid)
  g0[1] = ldsOff;                                    // lds_addr
  g0[2] = (unsigned)(ga & 0xFFFFFFFFu);              // global_addr lo
  g0[3] = (unsigned)((ga >> 32) & 0x01FFFFFFu) | (2u << 30); // addr hi | type=2
  tdm_v8i g1;
  g1[0] = (int)((1u << 16) | (1u << 20) | (5u << 22) | (3u << 25)); // 2B, pad en/int/amt
  g1[1] = (int)(128u << 16);   // tensor_dim0 = 128 (low16 at bit48)
  g1[2] = (int)(128u << 16);   // tensor_dim1 = 128 (low16 at bit80)
  g1[3] = (int)(128u << 16);   // tile_dim0   = 128 (bits 127:112)
  g1[4] = 128;                 // tile_dim1   = 128, tile_dim2 = 0
  g1[5] = 128;                 // tensor_dim0_stride = 128
  g1[6] = 0;                   // stride0 hi / stride1 lo
  g1[7] = 0;
  tdm_v4i gz = {0, 0, 0, 0};
#if __clang_major__ >= 23
  tdm_v8i gz8 = {0, 0, 0, 0, 0, 0, 0, 0};
  __builtin_amdgcn_tensor_load_to_lds(g0, g1, gz, gz, gz8, 0);
#else
  __builtin_amdgcn_tensor_load_to_lds(g0, g1, gz, gz, 0);
#endif
}
#endif

__global__ void k_fill(float* p, long long n, float v) {
  long long i = (long long)blockIdx.x * blockDim.x + threadIdx.x;
  if (i < n) p[i] = v;
}

__global__ void k_concat_cast(const float* h, const float* h0, float* x,
                              unsigned short* xbf, int N) {
  long long gid = (long long)blockIdx.x * blockDim.x + threadIdx.x;
  if (gid >= (long long)N * H) return;
  int node = (int)(gid >> 7);
  int c = (int)(gid & 127);
  float v = (c < 64) ? h[(long long)node * 64 + c]
                     : h0[(long long)node * 64 + (c - 64)];
  x[gid] = v;
  xbf[gid] = f2bf(v);
}

__global__ void k_deg(const int* dst, float* deg, int E) {
  int e = blockIdx.x * blockDim.x + threadIdx.x;
  if (e < E) atomicAdd(&deg[dst[e]], 1.0f);
}

// W is [k][n] row-major fp32; produce bf16 transposed Wt[n][k]
__global__ void k_wprep(const float* W, unsigned short* Wt) {
  int idx = blockIdx.x * blockDim.x + threadIdx.x; // = k*H + n
  int k = idx >> 7, n = idx & 127;
  Wt[n * H + k] = f2bf(W[idx]);
}

// one thread = one edge x 4 channels: float4 gather + 4 global f32 atomics
__global__ void k_scatter_mean(const float* __restrict__ x, const int* __restrict__ src,
                               const int* __restrict__ dst, const float* __restrict__ ew,
                               float* s, int E) {
  long long gid = (long long)blockIdx.x * blockDim.x + threadIdx.x;
  if (gid >= (long long)E * 32) return;
  int e = (int)(gid >> 5);
  int q = ((int)gid & 31) * 4;
  float w = ew[e];
  const float4 v = *(const float4*)(x + (long long)src[e] * H + q);
  float* sp = s + (long long)dst[e] * H + q;
  atomicAdd(sp + 0, v.x * w);
  atomicAdd(sp + 1, v.y * w);
  atomicAdd(sp + 2, v.z * w);
  atomicAdd(sp + 3, v.w * w);
}

__global__ void k_scatter_max(const float* __restrict__ p, const int* __restrict__ src,
                              const int* __restrict__ dst, const float* __restrict__ ew,
                              float* mx, int E) {
  long long gid = (long long)blockIdx.x * blockDim.x + threadIdx.x;
  if (gid >= (long long)E * 32) return;
  int e = (int)(gid >> 5);
  int q = ((int)gid & 31) * 4;
  float w = ew[e];
  const float4 v = *(const float4*)(p + (long long)src[e] * H + q);
  unsigned int* mp = (unsigned int*)(mx + (long long)dst[e] * H + q);
  // messages are relu(..)*uniform >= 0, so unsigned bit-pattern max == float max
  atomicMax(mp + 0, __float_as_uint(v.x * w));
  atomicMax(mp + 1, __float_as_uint(v.y * w));
  atomicMax(mp + 2, __float_as_uint(v.z * w));
  atomicMax(mp + 3, __float_as_uint(v.w * w));
}

__global__ void k_divdeg_cast(float* s, const float* deg, unsigned short* sbf, int N) {
  long long gid = (long long)blockIdx.x * blockDim.x + threadIdx.x;
  if (gid >= (long long)N * H) return;
  float d = deg[gid >> 7];
  d = d > 1.0f ? d : 1.0f;
  float v = s[gid] / d;
  s[gid] = v;
  sbf[gid] = f2bf(v);
}

__global__ void k_cast(const float* a, unsigned short* abf, long long n) {
  long long gid = (long long)blockIdx.x * blockDim.x + threadIdx.x;
  if (gid < n) abf[gid] = f2bf(a[gid]);
}

// per-channel sum and sum-of-squares via LDS atomics, then one global flush
__global__ void k_bn_stats(const float* __restrict__ y, float* stats, int N) {
  __shared__ float ls[256];
  ls[threadIdx.x] = 0.0f;
  __syncthreads();
  int c = threadIdx.x & 127;
  int rowsPerBlk = (N + gridDim.x - 1) / gridDim.x;
  int r0 = blockIdx.x * rowsPerBlk;
  int r1 = r0 + rowsPerBlk; if (r1 > N) r1 = N;
  for (int r = r0 + (threadIdx.x >> 7); r < r1; r += 2) {
    float v = y[(long long)r * H + c];
    atomicAdd(&ls[c], v);
    atomicAdd(&ls[128 + c], v * v);
  }
  __syncthreads();
  if (threadIdx.x < 128) {
    atomicAdd(&stats[threadIdx.x], ls[threadIdx.x]);
    atomicAdd(&stats[128 + threadIdx.x], ls[128 + threadIdx.x]);
  }
}

__global__ void k_bn_apply(float* y, const float* stats, const float* g,
                           const float* b, int N) {
  long long gid = (long long)blockIdx.x * blockDim.x + threadIdx.x;
  if (gid >= (long long)N * H) return;
  int c = (int)(gid & 127);
  float invN = 1.0f / (float)N;
  float mu = stats[c] * invN;
  float var = stats[128 + c] * invN - mu * mu;
  y[gid] = (y[gid] - mu) * rsqrtf(var + BN_EPS) * g[c] + b[c];
}

__global__ void k_combine(const float* o1, const float* o2, float* x,
                          unsigned short* xbf, long long n) {
  long long gid = (long long)blockIdx.x * blockDim.x + threadIdx.x;
  if (gid >= n) return;
  float v = o1[gid] + o2[gid];
  v = v > 0.0f ? v : LRELU_SLOPE * v;   // block-internal LeakyReLU
  v = v > 0.0f ? v : LRELU_SLOPE * v;   // ResSAGE per-block LeakyReLU
  x[gid] = v;
  xbf[gid] = f2bf(v);
}

// ---------------------------------------------------------------------------
// Fused dual GEMM: Y[N,H] = A1@W1 + A2@W2 (bf16 in, f32 out).
// One block = 64 rows x 128 cols; both 128x128 weight tiles TDM-DMA'd to LDS;
// 8 B fragments live in VGPRs; 8 WMMAs per row tile (32 static WMMAs).
// ---------------------------------------------------------------------------
__global__ __launch_bounds__(256) void k_gemm2(const unsigned short* __restrict__ A1bf,
                                               const unsigned short* __restrict__ A2bf,
                                               const unsigned short* __restrict__ W1t,
                                               const unsigned short* __restrict__ W2t,
                                               float* __restrict__ Y, int N) {
  __shared__ unsigned short W1Lds[H * LDSW];
  __shared__ unsigned short W2Lds[H * LDSW];

#if __has_builtin(__builtin_amdgcn_tensor_load_to_lds)
  if (threadIdx.x == 0) {
    tdm_load_weight(W1t, (unsigned)(unsigned long long)(&W1Lds[0]));
    tdm_load_weight(W2t, (unsigned)(unsigned long long)(&W2Lds[0]));
  }
  __builtin_amdgcn_s_wait_tensorcnt(0);
#else
  for (int i = threadIdx.x; i < H * H; i += 256) {
    W1Lds[(i >> 7) * LDSW + (i & 127)] = W1t[i];
    W2Lds[(i >> 7) * LDSW + (i & 127)] = W2t[i];
  }
#endif
  __syncthreads();

  const int wave = threadIdx.x >> 5;   // 8 waves -> 8 column tiles of 16
  const int lane = threadIdx.x & 31;
  const int m = lane & 15;
  const bool hi = lane >= 16;
  const int ncol = wave * 16 + m;
  const int kbA = hi ? 8 : 0;    // 16-bit A layout: lanes16-31 hold K+8 / K+24
  const int kbB = hi ? 8 : 0;    // 16-bit B layout: lanes16-31 hold K+8 / K+24
  const int mrow = hi ? 8 : 0;

  v16bf b1[4], b2[4];
#pragma unroll
  for (int kk = 0; kk < 4; ++kk) {
    const unsigned short* p1 = &W1Lds[ncol * LDSW + kk * 32 + kbB];
    const unsigned short* p2 = &W2Lds[ncol * LDSW + kk * 32 + kbB];
    v8bf lo1 = *(const v8bf*)p1, hi1 = *(const v8bf*)(p1 + 16);
    v8bf lo2 = *(const v8bf*)p2, hi2 = *(const v8bf*)(p2 + 16);
    b1[kk] = __builtin_shufflevector(lo1, hi1, 0, 1, 2, 3, 4, 5, 6, 7,
                                     8, 9, 10, 11, 12, 13, 14, 15);
    b2[kk] = __builtin_shufflevector(lo2, hi2, 0, 1, 2, 3, 4, 5, 6, 7,
                                     8, 9, 10, 11, 12, 13, 14, 15);
  }

  const long long blockRow0 = (long long)blockIdx.x * 64;
  const bool fullBlk = (blockRow0 + 64) <= (long long)N;
  if (fullBlk) {
    __builtin_prefetch(A1bf + (blockRow0 + 64) * H, 0, 0); // global_prefetch_b8
    __builtin_prefetch(A2bf + (blockRow0 + 64) * H, 0, 0);
  }

#pragma unroll
  for (int rt = 0; rt < 4; ++rt) {
    const long long row0 = blockRow0 + rt * 16;
    if (row0 < N) {
      int rowA = (int)row0 + m; if (rowA >= N) rowA = N - 1;
      const unsigned short* a1row = A1bf + (long long)rowA * H;
      const unsigned short* a2row = A2bf + (long long)rowA * H;

      v8f c = {0.f, 0.f, 0.f, 0.f, 0.f, 0.f, 0.f, 0.f};
#pragma unroll
      for (int kk = 0; kk < 4; ++kk) {
        const int k0 = kk * 32;
        v8bf alo = *(const v8bf*)(a1row + k0 + kbA);
        v8bf ahi = *(const v8bf*)(a1row + k0 + 16 + kbA);
        v16bf a = __builtin_shufflevector(alo, ahi, 0, 1, 2, 3, 4, 5, 6, 7,
                                          8, 9, 10, 11, 12, 13, 14, 15);
        c = __builtin_amdgcn_wmma_f32_16x16x32_bf16(
            false, a, false, b1[kk], (short)0, c, false, false);
      }
#pragma unroll
      for (int kk = 0; kk < 4; ++kk) {
        const int k0 = kk * 32;
        v8bf alo = *(const v8bf*)(a2row + k0 + kbA);
        v8bf ahi = *(const v8bf*)(a2row + k0 + 16 + kbA);
        v16bf a = __builtin_shufflevector(alo, ahi, 0, 1, 2, 3, 4, 5, 6, 7,
                                          8, 9, 10, 11, 12, 13, 14, 15);
        c = __builtin_amdgcn_wmma_f32_16x16x32_bf16(
            false, a, false, b2[kk], (short)0, c, false, false);
      }

      if (fullBlk) {
#pragma unroll
        for (int v = 0; v < 8; ++v)
          Y[(row0 + v + mrow) * H + ncol] = c[v];
      } else {
#pragma unroll
        for (int v = 0; v < 8; ++v) {
          long long r = row0 + v + mrow;
          if (r < N) Y[r * H + ncol] = c[v];
        }
      }
    }
  }
}

// ---------------------------------------------------------------------------
// Single GEMM with fused bias + relu: Y = relu(A@W + bias)  (pool fc_pool)
// ---------------------------------------------------------------------------
__global__ __launch_bounds__(256) void k_gemm_biasrelu(
    const unsigned short* __restrict__ Abf, const unsigned short* __restrict__ Wt,
    float* __restrict__ Y, const float* __restrict__ bias, int N) {
  __shared__ unsigned short WtLds[H * LDSW];

#if __has_builtin(__builtin_amdgcn_tensor_load_to_lds)
  if (threadIdx.x == 0)
    tdm_load_weight(Wt, (unsigned)(unsigned long long)(&WtLds[0]));
  __builtin_amdgcn_s_wait_tensorcnt(0);
#else
  for (int i = threadIdx.x; i < H * H; i += 256)
    WtLds[(i >> 7) * LDSW + (i & 127)] = Wt[i];
#endif
  __syncthreads();

  const int wave = threadIdx.x >> 5;
  const int lane = threadIdx.x & 31;
  const int m = lane & 15;
  const bool hi = lane >= 16;
  const int ncol = wave * 16 + m;
  const int kbA = hi ? 8 : 0;
  const int kbB = hi ? 8 : 0;
  const int mrow = hi ? 8 : 0;
  const float bcol = bias[ncol];

  v16bf bfrag[4];
#pragma unroll
  for (int kk = 0; kk < 4; ++kk) {
    const unsigned short* bp = &WtLds[ncol * LDSW + kk * 32 + kbB];
    v8bf blo = *(const v8bf*)bp;
    v8bf bhi = *(const v8bf*)(bp + 16);
    bfrag[kk] = __builtin_shufflevector(blo, bhi, 0, 1, 2, 3, 4, 5, 6, 7,
                                        8, 9, 10, 11, 12, 13, 14, 15);
  }

  const long long blockRow0 = (long long)blockIdx.x * 64;
  const bool fullBlk = (blockRow0 + 64) <= (long long)N;
  if (fullBlk)
    __builtin_prefetch(Abf + (blockRow0 + 64) * H, 0, 0);

#pragma unroll
  for (int rt = 0; rt < 4; ++rt) {
    const long long row0 = blockRow0 + rt * 16;
    if (row0 < N) {
      int rowA = (int)row0 + m; if (rowA >= N) rowA = N - 1;
      const unsigned short* arow = Abf + (long long)rowA * H;

      v8f c = {0.f, 0.f, 0.f, 0.f, 0.f, 0.f, 0.f, 0.f};
#pragma unroll
      for (int kk = 0; kk < 4; ++kk) {
        const int k0 = kk * 32;
        v8bf alo = *(const v8bf*)(arow + k0 + kbA);
        v8bf ahi = *(const v8bf*)(arow + k0 + 16 + kbA);
        v16bf a = __builtin_shufflevector(alo, ahi, 0, 1, 2, 3, 4, 5, 6, 7,
                                          8, 9, 10, 11, 12, 13, 14, 15);
        c = __builtin_amdgcn_wmma_f32_16x16x32_bf16(
            false, a, false, bfrag[kk], (short)0, c, false, false);
      }

      if (fullBlk) {
#pragma unroll
        for (int v = 0; v < 8; ++v) {
          float val = c[v] + bcol;
          val = val > 0.0f ? val : 0.0f;
          Y[(row0 + v + mrow) * H + ncol] = val;
        }
      } else {
#pragma unroll
        for (int v = 0; v < 8; ++v) {
          long long r = row0 + v + mrow;
          if (r < N) {
            float val = c[v] + bcol;
            val = val > 0.0f ? val : 0.0f;
            Y[r * H + ncol] = val;
          }
        }
      }
    }
  }
}

__global__ void k_final(const float* __restrict__ x, const float* __restrict__ hn,
                        const float* __restrict__ Wsf, const float* __restrict__ Wnf,
                        const float* __restrict__ bfp, float* outSig,
                        float* outLogit, int N) {
  int i = blockIdx.x * blockDim.x + threadIdx.x;
  if (i >= N) return;
  float acc = bfp[0];
  const float* xr = x + (long long)i * H;
  const float* hr = hn + (long long)i * H;
#pragma unroll 4
  for (int c = 0; c < H; ++c) acc += xr[c] * Wsf[c] + hr[c] * Wnf[c];
  outLogit[i] = acc;
  outSig[i] = 1.0f / (1.0f + __expf(-acc));
}

extern "C" void kernel_launch(void* const* d_in, const int* in_sizes, int n_in,
                              void* d_out, int out_size, void* d_ws, size_t ws_size,
                              hipStream_t stream) {
  (void)n_in; (void)out_size; (void)ws_size;
  const float* h   = (const float*)d_in[0];
  const float* h0  = (const float*)d_in[1];
  const int*   src = (const int*)d_in[2];
  const int*   dst = (const int*)d_in[3];
  const float* ew  = (const float*)d_in[4];
  const int N = in_sizes[0] / 64;
  const int E = in_sizes[2];
  const long long NH = (long long)N * H;

  // ---- workspace carve ----
  char* ws = (char*)d_ws;
  size_t off = 0;
  auto carve = [&](size_t bytes) -> void* {
    void* p = ws + off;
    off = (off + bytes + 255) & ~(size_t)255;
    return p;
  };
  float* x            = (float*)carve(NH * 4);
  unsigned short* xbf = (unsigned short*)carve(NH * 2);
  float* sbuf         = (float*)carve(NH * 4);   // segsum / h_neigh / segmax
  unsigned short* nbf = (unsigned short*)carve(NH * 2);
  float* out1         = (float*)carve(NH * 4);
  float* out2         = (float*)carve(NH * 4);
  float* pbuf         = (float*)carve(NH * 4);   // relu(x@Wp + bp)
  float* deg          = (float*)carve((size_t)N * 4);
  float* stats        = (float*)carve(256 * 4);
  unsigned short* Wt[10];
  for (int i = 0; i < 10; ++i) Wt[i] = (unsigned short*)carve((size_t)H * H * 2);

  const int T = 256;
  dim3 blk(T);
  auto g1 = [&](long long n) { return dim3((unsigned)((n + T - 1) / T)); };
  const dim3 gemmGrid((unsigned)((N + 63) / 64));

  // degree + input assembly
  k_fill<<<g1(N), blk, 0, stream>>>(deg, N, 0.0f);
  k_deg<<<g1(E), blk, 0, stream>>>(dst, deg, E);
  k_concat_cast<<<g1(NH), blk, 0, stream>>>(h, h0, x, xbf, N);

  // weight prep: per block b: Ws{b}a, Wn{b}a, Wp{b}, Ws{b}b, Wn{b}b
  const int wmap[10] = {5 + 0, 5 + 1, 5 + 4, 5 + 6, 5 + 7,
                        15 + 0, 15 + 1, 15 + 4, 15 + 6, 15 + 7};
  for (int i = 0; i < 10; ++i)
    k_wprep<<<dim3(64), blk, 0, stream>>>((const float*)d_in[wmap[i]], Wt[i]);

  for (int b = 0; b < 2; ++b) {
    const int base = 5 + b * 10;
    const float* ga = (const float*)d_in[base + 2];
    const float* ba = (const float*)d_in[base + 3];
    const float* bp = (const float*)d_in[base + 5];
    const float* gb = (const float*)d_in[base + 8];
    const float* bb = (const float*)d_in[base + 9];
    unsigned short* Wsa = Wt[b * 5 + 0];
    unsigned short* Wna = Wt[b * 5 + 1];
    unsigned short* Wp  = Wt[b * 5 + 2];
    unsigned short* Wsb = Wt[b * 5 + 3];
    unsigned short* Wnb = Wt[b * 5 + 4];

    // ---- mean branch: out1 = x@Wsa + (segsum(x[src]*ew)/deg)@Wna ; BN ----
    k_fill<<<g1(NH), blk, 0, stream>>>(sbuf, NH, 0.0f);
    k_scatter_mean<<<g1((long long)E * 32), blk, 0, stream>>>(x, src, dst, ew, sbuf, E);
    k_divdeg_cast<<<g1(NH), blk, 0, stream>>>(sbuf, deg, nbf, N);
    k_gemm2<<<gemmGrid, blk, 0, stream>>>(xbf, nbf, Wsa, Wna, out1, N);
    k_fill<<<dim3(1), blk, 0, stream>>>(stats, 256, 0.0f);
    k_bn_stats<<<dim3(512), blk, 0, stream>>>(out1, stats, N);
    k_bn_apply<<<g1(NH), blk, 0, stream>>>(out1, stats, ga, ba, N);

    // ---- pool branch: p = relu(x@Wp+bp); out2 = x@Wsb + segmax(p[src]*ew)@Wnb; BN
    k_gemm_biasrelu<<<gemmGrid, blk, 0, stream>>>(xbf, Wp, pbuf, bp, N);
    k_fill<<<g1(NH), blk, 0, stream>>>(sbuf, NH, 0.0f);  // 0-init = zero-degree case
    k_scatter_max<<<g1((long long)E * 32), blk, 0, stream>>>(pbuf, src, dst, ew, sbuf, E);
    k_cast<<<g1(NH), blk, 0, stream>>>(sbuf, nbf, NH);
    k_gemm2<<<gemmGrid, blk, 0, stream>>>(xbf, nbf, Wsb, Wnb, out2, N);
    k_fill<<<dim3(1), blk, 0, stream>>>(stats, 256, 0.0f);
    k_bn_stats<<<dim3(512), blk, 0, stream>>>(out2, stats, N);
    k_bn_apply<<<g1(NH), blk, 0, stream>>>(out2, stats, gb, bb, N);

    k_combine<<<g1(NH), blk, 0, stream>>>(out1, out2, x, xbf, NH);
  }

  // ---- final mean-aggregate + dot products + sigmoid ----
  k_fill<<<g1(NH), blk, 0, stream>>>(sbuf, NH, 0.0f);
  k_scatter_mean<<<g1((long long)E * 32), blk, 0, stream>>>(x, src, dst, ew, sbuf, E);
  k_divdeg_cast<<<g1(NH), blk, 0, stream>>>(sbuf, deg, nbf, N);
  float* outSig = (float*)d_out;
  float* outLog = outSig + N;
  k_final<<<g1(N), blk, 0, stream>>>(x, sbuf, (const float*)d_in[25],
                                     (const float*)d_in[26], (const float*)d_in[27],
                                     outSig, outLog, N);
}
